// SelfAttention_44890998178140
// MI455X (gfx1250) — matrix operand order repaired and checked by reading, not compiled, not run on hardware
//
#include <hip/hip_runtime.h>

typedef _Float16 h16;
typedef __attribute__((ext_vector_type(16))) _Float16 v16h;
typedef __attribute__((ext_vector_type(8)))  _Float16 v8h;
typedef __attribute__((ext_vector_type(4)))  _Float16 v4h;
typedef __attribute__((ext_vector_type(8)))  float    v8f;
typedef __attribute__((ext_vector_type(4)))  float    v4f;

#define TM 128            // block tile M
#define TN 128            // block tile N
#define TK 32             // K slab
#define PAD 8
#define LROW (TK + PAD)   // padded LDS row stride: 80B = 5*16B, keeps 16B align

union FragU { v16h v; v8h h[2]; };

// ---------------------------------------------------------------- f32 -> f16
__global__ __launch_bounds__(256)
void attn_cvt_f32_f16_x4(const float* __restrict__ s, h16* __restrict__ d, int n4) {
    int i = blockIdx.x * blockDim.x + threadIdx.x;
    if (i < n4) {
        v4f f = ((const v4f*)s)[i];
        v4h h;
        h.x = (h16)f.x; h.y = (h16)f.y; h.z = (h16)f.z; h.w = (h16)f.w;
        ((v4h*)d)[i] = h;
    }
}

// ------------------------------------------------- C[m,n] = sum_k A[m,k]*B[n,k]
// A: M x K row-major, B: N x K row-major (NT gemm: x@W^T and q@k^T).
// Grid: (N/TN, M/TM, batch). 256 threads = 8 waves arranged 4(m) x 2(n);
// each wave computes 32x64 = 2 A-frags x 4 B-frags = 8 WMMAs per K slab.
// All fragments are preloaded before the WMMA burst; global loads for slab
// k+1 are issued into registers before computing slab k.
template <bool OUT_F32>
__global__ __launch_bounds__(256)
void attn_gemm_nt_wmma(const h16* __restrict__ A, const h16* __restrict__ Bm,
                       void* __restrict__ Cout,
                       int M, int N, int K,
                       long long sA, long long sB, long long sC)
{
    __shared__ __align__(16) h16 As[TM * LROW];
    __shared__ __align__(16) h16 Bs[TN * LROW];
    (void)M;

    const int b = blockIdx.z;
    const h16* Ab = A  + (size_t)b * sA;
    const h16* Bb = Bm + (size_t)b * sB;
    const int m0 = blockIdx.y * TM;
    const int n0 = blockIdx.x * TN;
    const int tid  = threadIdx.x;
    const int wave = tid >> 5;
    const int lane = tid & 31;
    const int lm   = lane & 15;         // fragment row (A) / col (B)
    const int lh   = lane >> 4;         // half-wave selector
    const int wm   = (wave & 3) * 32;   // wave M offset in tile
    const int wn   = (wave >> 2) * 64;  // wave N offset in tile

    v8f acc[2][4] = {};
    v8h ar[2], br[2];

    // 128 rows x 4 chunks(16B) per tile = 512 chunks; 2 per thread.
    auto load_tiles = [&](int kk) {
        #pragma unroll
        for (int c = 0; c < 2; ++c) {
            int chunk = tid + c * 256;
            int r  = chunk >> 2;
            int cc = (chunk & 3) << 3;
            ar[c] = *(const v8h*)(Ab + (size_t)(m0 + r) * K + kk + cc);
            br[c] = *(const v8h*)(Bb + (size_t)(n0 + r) * K + kk + cc);
        }
    };
    auto store_tiles = [&]() {
        #pragma unroll
        for (int c = 0; c < 2; ++c) {
            int chunk = tid + c * 256;
            int r  = chunk >> 2;
            int cc = (chunk & 3) << 3;
            *(v8h*)&As[r * LROW + cc] = ar[c];
            *(v8h*)&Bs[r * LROW + cc] = br[c];
        }
    };

    load_tiles(0);
    store_tiles();
    __syncthreads();

    for (int kk = 0; kk < K; kk += TK) {
        const bool more = (kk + TK) < K;
        if (more) load_tiles(kk + TK);          // overlap with WMMA below
        if (kk + 2 * TK < K) {                  // L2 prefetch two slabs ahead
            int r = tid & 127;
            const h16* pf = (tid < 128) ? (Ab + (size_t)(m0 + r) * K + kk + 2 * TK)
                                        : (Bb + (size_t)(n0 + r) * K + kk + 2 * TK);
            __builtin_prefetch(pf, 0, 1);
        }

        // Preload ALL fragments, then fire the 8-WMMA burst: batches the
        // ds_load_b128s behind one s_wait_dscnt instead of one wait per pair.
        FragU a[2], bf[4];
        #pragma unroll
        for (int i = 0; i < 2; ++i) {
            // lane lm = row, half-wave lh: K = lh*8 + {0..7, 16..23}
            const h16* ap = &As[(wm + i * 16 + lm) * LROW + lh * 8];
            a[i].h[0] = *(const v8h*)(ap);
            a[i].h[1] = *(const v8h*)(ap + 16);
        }
        #pragma unroll
        for (int j = 0; j < 4; ++j) {
            // lane lm = col, half-wave lh: K = lh*8 + {0..7, 16..23}
            const h16* bp = &Bs[(wn + j * 16 + lm) * LROW + lh * 8];
            bf[j].h[0] = *(const v8h*)(bp);
            bf[j].h[1] = *(const v8h*)(bp + 16);
        }
        #pragma unroll
        for (int j = 0; j < 4; ++j)
            #pragma unroll
            for (int i = 0; i < 2; ++i)
                acc[i][j] = __builtin_amdgcn_wmma_f32_16x16x32_f16(
                    false, a[i].v, false, bf[j].v, (short)0, acc[i][j], false, false);

        __syncthreads();
        if (more) { store_tiles(); __syncthreads(); }
    }

    // D layout: VGPR r -> row (r + lh*8), lane lm -> col.
    #pragma unroll
    for (int i = 0; i < 2; ++i) {
        const int row0 = m0 + wm + i * 16 + lh * 8;
        #pragma unroll
        for (int j = 0; j < 4; ++j) {
            const int col = n0 + wn + j * 16 + lm;
            if constexpr (OUT_F32) {
                float* C = (float*)Cout + (size_t)b * sC;
                #pragma unroll
                for (int r = 0; r < 8; ++r)
                    C[(size_t)(row0 + r) * N + col] = acc[i][j][r];
            } else {
                h16* C = (h16*)Cout + (size_t)b * sC;
                #pragma unroll
                for (int r = 0; r < 8; ++r)
                    C[(size_t)(row0 + r) * N + col] = (h16)acc[i][j][r];
            }
        }
    }
}

// ------------------------------------------------- C[m,n] = sum_k A[m,k]*B[k,n]
// A: M x K row-major (attn probs), B: K x N row-major (V). B staged transposed
// into LDS (coalesced global reads, scattered ds stores). Same 8-wave pipeline.
__global__ __launch_bounds__(256)
void attn_gemm_nn_wmma(const h16* __restrict__ A, const h16* __restrict__ Bm,
                       float* __restrict__ Cout,
                       int M, int N, int K,
                       long long sA, long long sB, long long sC)
{
    __shared__ __align__(16) h16 As[TM * LROW];
    __shared__ __align__(16) h16 Bs[TN * LROW];   // Bs[n][k] = B[kk+k][n0+n]
    (void)M;

    const int b = blockIdx.z;
    const h16* Ab = A  + (size_t)b * sA;
    const h16* Bb = Bm + (size_t)b * sB;
    const int m0 = blockIdx.y * TM;
    const int n0 = blockIdx.x * TN;
    const int tid  = threadIdx.x;
    const int wave = tid >> 5;
    const int lane = tid & 31;
    const int lm   = lane & 15;
    const int lh   = lane >> 4;
    const int wm   = (wave & 3) * 32;
    const int wn   = (wave >> 2) * 64;

    v8f acc[2][4] = {};
    v8h ar[2], br[2];

    auto load_tiles = [&](int kk) {
        #pragma unroll
        for (int c = 0; c < 2; ++c) {
            int chunk = tid + c * 256;
            // A: 128 rows x 4 chunks
            int r  = chunk >> 2;
            int cc = (chunk & 3) << 3;
            ar[c] = *(const v8h*)(Ab + (size_t)(m0 + r) * K + kk + cc);
            // B: 32 k-rows x 16 n-chunks (coalesced along n)
            int kr = chunk >> 4;
            int nc = (chunk & 15) << 3;
            br[c] = *(const v8h*)(Bb + (size_t)(kk + kr) * N + n0 + nc);
        }
    };
    auto store_tiles = [&]() {
        #pragma unroll
        for (int c = 0; c < 2; ++c) {
            int chunk = tid + c * 256;
            int r  = chunk >> 2;
            int cc = (chunk & 3) << 3;
            *(v8h*)&As[r * LROW + cc] = ar[c];
            int kr = chunk >> 4;
            int nc = (chunk & 15) << 3;
            #pragma unroll
            for (int j = 0; j < 8; ++j)
                Bs[(nc + j) * LROW + kr] = br[c][j];
        }
    };

    load_tiles(0);
    store_tiles();
    __syncthreads();

    for (int kk = 0; kk < K; kk += TK) {
        const bool more = (kk + TK) < K;
        if (more) load_tiles(kk + TK);
        if (kk + 2 * TK < K) {
            if (tid < 128) {
                __builtin_prefetch(Ab + (size_t)(m0 + tid) * K + kk + 2 * TK, 0, 1);
            } else if (tid < 160) {
                __builtin_prefetch(Bb + (size_t)(kk + 2 * TK + (tid - 128)) * N + n0, 0, 1);
            }
        }

        FragU a[2], bf[4];
        #pragma unroll
        for (int i = 0; i < 2; ++i) {
            const h16* ap = &As[(wm + i * 16 + lm) * LROW + lh * 8];
            a[i].h[0] = *(const v8h*)(ap);
            a[i].h[1] = *(const v8h*)(ap + 16);
        }
        #pragma unroll
        for (int j = 0; j < 4; ++j) {
            const h16* bp = &Bs[(wn + j * 16 + lm) * LROW + lh * 8];
            bf[j].h[0] = *(const v8h*)(bp);
            bf[j].h[1] = *(const v8h*)(bp + 16);
        }
        #pragma unroll
        for (int j = 0; j < 4; ++j)
            #pragma unroll
            for (int i = 0; i < 2; ++i)
                acc[i][j] = __builtin_amdgcn_wmma_f32_16x16x32_f16(
                    false, a[i].v, false, bf[j].v, (short)0, acc[i][j], false, false);

        __syncthreads();
        if (more) { store_tiles(); __syncthreads(); }
    }

    #pragma unroll
    for (int i = 0; i < 2; ++i) {
        const int row0 = m0 + wm + i * 16 + lh * 8;
        #pragma unroll
        for (int j = 0; j < 4; ++j) {
            const int col = n0 + wn + j * 16 + lm;
            float* C = Cout + (size_t)b * sC;
            #pragma unroll
            for (int r = 0; r < 8; ++r)
                C[(size_t)(row0 + r) * N + col] = acc[i][j][r];
        }
    }
}

// ---------------------------------------------------- row softmax (len 2048)
__global__ __launch_bounds__(256)
void attn_softmax_rows(const float* __restrict__ S, h16* __restrict__ P, float scale)
{
    __shared__ float red[256];
    const size_t row = blockIdx.x;
    const float* s = S + row * 2048;
    h16*         p = P + row * 2048;
    const int t = threadIdx.x;

    float v[8];
    float mx = -__builtin_inff();
    #pragma unroll
    for (int i = 0; i < 8; ++i) { v[i] = s[t + i * 256] * scale; mx = fmaxf(mx, v[i]); }
    red[t] = mx; __syncthreads();
    for (int w = 128; w > 0; w >>= 1) { if (t < w) red[t] = fmaxf(red[t], red[t + w]); __syncthreads(); }
    mx = red[0]; __syncthreads();

    float sum = 0.f;
    #pragma unroll
    for (int i = 0; i < 8; ++i) { v[i] = __expf(v[i] - mx); sum += v[i]; }
    red[t] = sum; __syncthreads();
    for (int w = 128; w > 0; w >>= 1) { if (t < w) red[t] += red[t + w]; __syncthreads(); }
    const float inv = 1.0f / red[0];
    #pragma unroll
    for (int i = 0; i < 8; ++i) p[t + i * 256] = (h16)(v[i] * inv);
}

// ---------------------------------------------------------------------------
extern "C" void kernel_launch(void* const* d_in, const int* in_sizes, int n_in,
                              void* d_out, int out_size, void* d_ws, size_t ws_size,
                              hipStream_t stream) {
    (void)in_sizes; (void)n_in; (void)out_size; (void)ws_size;
    const float* x  = (const float*)d_in[0];
    const float* Wq = (const float*)d_in[1];
    const float* Wk = (const float*)d_in[2];
    const float* Wv = (const float*)d_in[3];
    float* out = (float*)d_out;
    char* ws = (char*)d_ws;

    const size_t NX = (size_t)8192 * 1024;      // x / q / k / v elements
    const size_t NW = (size_t)1024 * 1024;      // weight elements
    const size_t NS = (size_t)4 * 2048 * 2048;  // scores elements

    size_t off = 0;
    auto take = [&](size_t bytes) { size_t o = off; off += (bytes + 255) & ~(size_t)255; return o; };
    h16*   xh  = (h16*)  (ws + take(NX * 2));
    h16*   wqh = (h16*)  (ws + take(NW * 2));
    h16*   wkh = (h16*)  (ws + take(NW * 2));
    h16*   wvh = (h16*)  (ws + take(NW * 2));
    h16*   qh  = (h16*)  (ws + take(NX * 2));
    h16*   kh  = (h16*)  (ws + take(NX * 2));
    h16*   vh  = (h16*)  (ws + take(NX * 2));
    float* sc  = (float*)(ws + take(NS * 4));
    h16*   ph  = (h16*)  (ws + take(NS * 2));

    // 1) cast to f16
    {
        int n4 = (int)(NX / 4);
        attn_cvt_f32_f16_x4<<<(n4 + 255) / 256, 256, 0, stream>>>(x, xh, n4);
        int w4 = (int)(NW / 4);
        attn_cvt_f32_f16_x4<<<(w4 + 255) / 256, 256, 0, stream>>>(Wq, wqh, w4);
        attn_cvt_f32_f16_x4<<<(w4 + 255) / 256, 256, 0, stream>>>(Wk, wkh, w4);
        attn_cvt_f32_f16_x4<<<(w4 + 255) / 256, 256, 0, stream>>>(Wv, wvh, w4);
    }

    // 2) q/k/v = x @ W^T   (M=8192, N=1024, K=1024)
    {
        dim3 g(1024 / TN, 8192 / TM, 1);
        attn_gemm_nt_wmma<false><<<g, 256, 0, stream>>>(xh, wqh, qh, 8192, 1024, 1024, 0, 0, 0);
        attn_gemm_nt_wmma<false><<<g, 256, 0, stream>>>(xh, wkh, kh, 8192, 1024, 1024, 0, 0, 0);
        attn_gemm_nt_wmma<false><<<g, 256, 0, stream>>>(xh, wvh, vh, 8192, 1024, 1024, 0, 0, 0);
    }

    // 3) scores = q @ k^T per batch (M=N=2048, K=1024), f32 out
    {
        dim3 g(2048 / TN, 2048 / TM, 4);
        attn_gemm_nt_wmma<true><<<g, 256, 0, stream>>>(
            qh, kh, sc, 2048, 2048, 1024,
            (long long)2048 * 1024, (long long)2048 * 1024, (long long)2048 * 2048);
    }

    // 4) softmax(scores / 32) -> f16 probs
    attn_softmax_rows<<<4 * 2048, 256, 0, stream>>>(sc, ph, 0.03125f);

    // 5) out = probs @ v per batch (M=2048, N=1024, K=2048), f32 out
    {
        dim3 g(1024 / TN, 2048 / TM, 4);
        attn_gemm_nn_wmma<<<g, 256, 0, stream>>>(
            ph, vh, out, 2048, 1024, 2048,
            (long long)2048 * 2048, (long long)2048 * 1024, (long long)2048 * 1024);
    }
}
